// GeneralizedRingAttractor_74388833566851
// MI455X (gfx1250) — matrix operand order repaired and checked by reading, not compiled, not run on hardware
//
#include <hip/hip_runtime.h>
#include <hip/hip_bf16.h>
#include <math.h>

typedef __attribute__((ext_vector_type(16))) __bf16 v16bf;
typedef __attribute__((ext_vector_type(8)))  __bf16 v8bf;
typedef __attribute__((ext_vector_type(8)))  float  v8f;

#define Bn 64
#define Tn 512
#define Nn 512
#define Kn 2
#define Hn 16
#define KEXT (3 * Nn)          // 1536: [J1*Wo ; Wa0 ; Wa1] stacked along K
#define NKC  (KEXT / 32)       // 48 K-chunks of 32
#define NGROUP (Bn / 16)       // 4 batch groups
#define NSLICE 8               // workgroups per group (each owns 64 rows of N)
#define TPB    128             // 4 waves per workgroup

constexpr float ALPHA  = 0.15f;
constexpr float J0_VAL = -0.1f;
constexpr float J1_VAL = 0.1f;
constexpr float TWO_PI_F = 6.28318530717958647692f;
constexpr float PI_F     = 3.14159265358979323846f;

// workspace layout (bytes)
#define WS_WCAT  0
#define WS_AG    (WS_WCAT + 3 * Nn * Nn * 2)            // 1,572,864
#define WS_STAGE (WS_AG + Bn * Tn * Kn * 4)             // +262,144
#define WS_GS    (WS_STAGE + NGROUP * 2 * 16 * KEXT * 2)// +393,216
#define STATN    (NGROUP * (Tn + 1) * 16)               // per stats array
#define WS_GC    (WS_GS + STATN * 4)
#define WS_GSN   (WS_GC + STATN * 4)
#define WS_CNT   (WS_GSN + STATN * 4)
#define ZERO_INTS (3 * STATN + NGROUP * Tn)             // stats + counters

__device__ __forceinline__ float fast_tanh(float x) {
    float e = __expf(-2.f * fabsf(x));
    return copysignf((1.f - e) / (1.f + e), x);
}

// ---------------------------------------------------------------------------
// Pre-pass 1: repack [J1*Wo ; Wa0 ; Wa1] (K-concatenated) into bf16 WMMA
// A-tiles. Tile (jn, kc) is 16 rows (n) x 32 cols (kext): 32 lanes x 16 bf16
// in contiguous k order, k = 16*(lane/16) + e (one 32B load/lane/tile).
// ---------------------------------------------------------------------------
__global__ void prep_weights(const float* __restrict__ Wo,
                             const float* __restrict__ Wa,
                             __bf16* __restrict__ wcat) {
    int id   = blockIdx.x * 256 + threadIdx.x;   // 0 .. 3*N*N-1
    int e    = id & 15;
    int lane = (id >> 4) & 31;
    int tile = id >> 9;
    int kc   = tile % NKC;
    int jn   = tile / NKC;

    int M, K;
    if (lane < 16) { M = lane;      K = e; }
    else           { M = lane - 16; K = 16 + e; }
    int n    = jn * 16 + M;
    int kext = kc * 32 + K;
    int w    = kext >> 9;
    int m    = kext & (Nn - 1);

    const float* src = (w == 0) ? Wo : (Wa + (size_t)(w - 1) * Nn * Nn);
    float v = src[(size_t)n * Nn + m];
    if (w == 0) v *= J1_VAL;
    wcat[id] = (__bf16)v;
}

// ---------------------------------------------------------------------------
// Pre-pass 2: A[b,t,k] = action * softplus(W2 . gelu(|a|*W1 + b1) + b2)
// ---------------------------------------------------------------------------
__global__ void prep_gains(const float* __restrict__ act,
                           const float* __restrict__ gw1,
                           const float* __restrict__ gb1,
                           const float* __restrict__ gw2,
                           const float* __restrict__ gb2,
                           float* __restrict__ Ag) {
    int id = blockIdx.x * 256 + threadIdx.x;
    int k = id & (Kn - 1);
    float a = act[id];
    float x = fabsf(a);
    float acc = 0.f;
#pragma unroll
    for (int h = 0; h < Hn; ++h) {
        float z = x * gw1[k * Hn + h] + gb1[k * Hn + h];
        float g = 0.5f * z * (1.0f + erff(z * 0.70710678118654752440f));
        acc += g * gw2[k * Hn + h];
    }
    acc += gb2[k];
    float sp = (acc > 20.f) ? acc : log1pf(expf(acc));
    Ag[id] = a * sp;
}

// Pre-pass 3: zero per-step stats + barrier counters (graph-replay safe)
__global__ void prep_zero(int* __restrict__ p) {
    int id = blockIdx.x * 256 + threadIdx.x;
    if (id < ZERO_INTS) p[id] = 0;
}

// ---------------------------------------------------------------------------
// Main recurrence: NGROUP*NSLICE = 32 workgroups, 4 waves each.
// Workgroup (g, sl) owns rows [sl*64, sl*64+64) for batch group g; wave wv
// owns one 16-row n-tile. Weight traffic: 192 KB/WG/step from L2. After each
// step the 8 slices exchange r via an L2 staging buffer + counter barrier
// with agent-scope release/acquire fences.
// ---------------------------------------------------------------------------
__global__ __launch_bounds__(TPB) void ring_recur(
        const __bf16* __restrict__ wcat,
        const float*  __restrict__ Ag,
        __bf16* __restrict__ stage,   // [NGROUP][2][16][KEXT]
        float*  __restrict__ gS,      // [NGROUP][Tn+1][16]  sum(r_t)
        float*  __restrict__ gC,      // [NGROUP][Tn+1][16]  <cos,r_t>
        float*  __restrict__ gSn,     // [NGROUP][Tn+1][16]  <sin,r_t>
        int*    __restrict__ gcnt,    // [NGROUP][Tn]
        float*  __restrict__ out_rd7,
        float*  __restrict__ out_r) {

    __shared__ __align__(32) __bf16 rbf[16 * KEXT];    // 48 KB extended B
    __shared__ float cth[Nn], sth[Nn];
    __shared__ float s0loc;

    const int tid   = threadIdx.x;
    const int lane  = tid & 31;
    const int wv    = tid >> 5;               // wave 0..3
    const int b     = lane & 15;
    const int hi    = lane >> 4;
    const int g     = blockIdx.x >> 3;        // batch group
    const int sl    = blockIdx.x & 7;         // N slice
    const int jn    = sl * 4 + wv;            // this wave's n-tile
    const int nb    = jn * 16 + 8 * hi;       // lane's first row
    const int bglob = g * 16 + b;

    for (int n = tid; n < Nn; n += TPB) {
        float th = TWO_PI_F * (float)n / (float)Nn;
        cth[n] = cosf(th);
        sth[n] = sinf(th);
    }
    if (tid == 0) s0loc = 0.f;
    __syncthreads();

    // ---- r0 bump (grid hits theta=pi exactly -> max-normalizer == 1) ------
    float r[8];
    {
        float part = 0.f;
#pragma unroll
        for (int q = 0; q < 4; ++q) {
            int n = tid * 4 + q;
            part += expf(2.f * (cosf(TWO_PI_F * (float)n / (float)Nn - PI_F) - 1.f));
        }
        atomicAdd(&s0loc, part);
#pragma unroll
        for (int v = 0; v < 8; ++v)
            r[v] = expf(2.f * (cosf(TWO_PI_F * (float)(nb + v) / (float)Nn - PI_F) - 1.f));
        // full initial extended B built locally (r0 is analytic, all rows)
        for (int idx = tid; idx < 16 * KEXT; idx += TPB) {
            int b2   = idx / KEXT;
            int kext = idx - b2 * KEXT;
            int w    = kext >> 9;
            int m    = kext & (Nn - 1);
            float val = expf(2.f * (cosf(TWO_PI_F * (float)m / (float)Nn - PI_F) - 1.f));
            if (w > 0) val *= Ag[((size_t)(g * 16 + b2) * Tn) * 2 + (w - 1)];
            rbf[idx] = (__bf16)val;
        }
    }
    __syncthreads();

    const __bf16* wl = wcat + (((size_t)jn * NKC) << 9) + (size_t)lane * 16;

    for (int t = 0; t < Tn; ++t) {
        // -------- phase A: 16-row GEMM over K=1536, 4 interleaved chains ---
        float S0;
        if (t == 0) S0 = J0_VAL * s0loc;
        else        S0 = J0_VAL * gS[((size_t)g * (Tn + 1) + t) * 16 + b];

        v8f a0 = (v8f){}, a1 = (v8f){}, a2 = (v8f){}, a3 = (v8f){};
        for (int kc = 0; kc < NKC; kc += 4) {
            const __bf16* lb = &rbf[b * KEXT + kc * 32 + 16 * hi];
            const v16bf bm0 = *(const v16bf*)(lb);
            const v16bf bm1 = *(const v16bf*)(lb + 32);
            const v16bf bm2 = *(const v16bf*)(lb + 64);
            const v16bf bm3 = *(const v16bf*)(lb + 96);
            const v16bf wa0 = *(const v16bf*)(wl + ((size_t)(kc + 0) << 9));
            const v16bf wa1 = *(const v16bf*)(wl + ((size_t)(kc + 1) << 9));
            const v16bf wa2 = *(const v16bf*)(wl + ((size_t)(kc + 2) << 9));
            const v16bf wa3 = *(const v16bf*)(wl + ((size_t)(kc + 3) << 9));
            a0 = __builtin_amdgcn_wmma_f32_16x16x32_bf16(false, wa0, false, bm0, (short)0, a0, false, false);
            a1 = __builtin_amdgcn_wmma_f32_16x16x32_bf16(false, wa1, false, bm1, (short)0, a1, false, false);
            a2 = __builtin_amdgcn_wmma_f32_16x16x32_bf16(false, wa2, false, bm2, (short)0, a2, false, false);
            a3 = __builtin_amdgcn_wmma_f32_16x16x32_bf16(false, wa3, false, bm3, (short)0, a3, false, false);
        }
        const v8f acc = (a0 + a1) + (a2 + a3);

#pragma unroll
        for (int v = 0; v < 8; ++v) {
            float rec = fast_tanh(S0 + acc[v]);
            r[v] = r[v] * (1.f - ALPHA) + ALPHA * rec;
        }

        // -------- phase B: publish slice, stats, history --------------------
        const int tn1 = (t + 1 < Tn) ? (t + 1) : (Tn - 1);
        const float2 aN = *(const float2*)(Ag + ((size_t)bglob * Tn + tn1) * 2);

        v8bf p0, p1, p2;
        float sp = 0.f, cp = 0.f, snp = 0.f;
#pragma unroll
        for (int v = 0; v < 8; ++v) {
            float val = r[v];
            p0[v] = (__bf16)val;
            p1[v] = (__bf16)(aN.x * val);
            p2[v] = (__bf16)(aN.y * val);
            sp  += val;
            cp  += cth[nb + v] * val;
            snp += sth[nb + v] * val;
        }
        __bf16* sb = stage + (((size_t)g * 2 + ((t + 1) & 1)) * 16 + b) * KEXT + nb;
        *(v8bf*)(sb)          = p0;
        *(v8bf*)(sb + Nn)     = p1;
        *(v8bf*)(sb + 2 * Nn) = p2;

        float* dst = out_r + ((size_t)bglob * Tn + t) * Nn + nb;
        *(float4*)(dst)     = make_float4(r[0], r[1], r[2], r[3]);
        *(float4*)(dst + 4) = make_float4(r[4], r[5], r[6], r[7]);

        sp  += __shfl_down(sp, 16, 32);          // fold lane+16 (same b)
        cp  += __shfl_down(cp, 16, 32);
        snp += __shfl_down(snp, 16, 32);
        const size_t si = ((size_t)g * (Tn + 1) + (t + 1)) * 16 + b;
        if (lane < 16) {
            atomicAdd(&gS[si],  sp);
            atomicAdd(&gC[si],  cp);
            atomicAdd(&gSn[si], snp);
        }

        // -------- inter-slice barrier (release -> count -> acquire) --------
        __builtin_amdgcn_fence(__ATOMIC_RELEASE, "agent");  // flush this wave's stores
        __syncthreads();
        if (tid == 0) {
            __hip_atomic_fetch_add(&gcnt[g * Tn + t], 1,
                                   __ATOMIC_RELEASE, __HIP_MEMORY_SCOPE_AGENT);
            while (__hip_atomic_load(&gcnt[g * Tn + t],
                                     __ATOMIC_ACQUIRE, __HIP_MEMORY_SCOPE_AGENT) < NSLICE)
                __builtin_amdgcn_s_sleep(2);
        }
        __syncthreads();
        __builtin_amdgcn_fence(__ATOMIC_ACQUIRE, "agent"); // invalidate stale L0

        // -------- copy full extended B (48 KB) from staging into LDS -------
        {
            const uint4* src4 = (const uint4*)(stage + ((size_t)g * 2 + ((t + 1) & 1)) * 16 * KEXT);
            uint4* dst4 = (uint4*)rbf;
            for (int i = tid; i < (16 * KEXT) / 8; i += TPB)  // 3072 uint4
                dst4[i] = src4[i];
        }

        // -------- phase C: rd7 via rank-2 W_delta7 + closed-form grid max --
        const float c = gC[si - b + b], s = gSn[si];
        const float cc = gC[si];
        const float phi = atan2f(s, cc);
        const float kf  = rintf(phi * ((float)Nn / TWO_PI_F));
        const float thk = kf * (TWO_PI_F / (float)Nn);
        const float inv = 1.0f / (cc * cosf(thk) + s * sinf(thk));
        float o[8];
#pragma unroll
        for (int v = 0; v < 8; ++v)
            o[v] = (cc * cth[nb + v] + s * sth[nb + v]) * inv;
        float* dq = out_rd7 + ((size_t)bglob * Tn + t) * Nn + nb;
        *(float4*)(dq)     = make_float4(o[0], o[1], o[2], o[3]);
        *(float4*)(dq + 4) = make_float4(o[4], o[5], o[6], o[7]);

        __syncthreads();   // LDS copy complete before next GEMM reads rbf
    }
}

// ---------------------------------------------------------------------------
extern "C" void kernel_launch(void* const* d_in, const int* in_sizes, int n_in,
                              void* d_out, int out_size, void* d_ws, size_t ws_size,
                              hipStream_t stream) {
    const float* act = (const float*)d_in[0];
    const float* Wo  = (const float*)d_in[1];
    const float* Wa  = (const float*)d_in[2];
    const float* gw1 = (const float*)d_in[3];
    const float* gb1 = (const float*)d_in[4];
    const float* gw2 = (const float*)d_in[5];
    const float* gb2 = (const float*)d_in[6];

    char* ws = (char*)d_ws;
    __bf16* wcat  = (__bf16*)(ws + WS_WCAT);
    float*  Agd   = (float*)(ws + WS_AG);
    __bf16* stg   = (__bf16*)(ws + WS_STAGE);
    float*  gS    = (float*)(ws + WS_GS);
    float*  gC    = (float*)(ws + WS_GC);
    float*  gSn   = (float*)(ws + WS_GSN);
    int*    gcnt  = (int*)(ws + WS_CNT);

    float* out_rd7 = (float*)d_out;
    float* out_r   = out_rd7 + (size_t)Bn * Tn * Nn;

    prep_weights<<<(3 * Nn * Nn) / 256, 256, 0, stream>>>(Wo, Wa, wcat);
    prep_gains<<<(Bn * Tn * Kn) / 256, 256, 0, stream>>>(act, gw1, gb1, gw2, gb2, Agd);
    prep_zero<<<(ZERO_INTS + 255) / 256, 256, 0, stream>>>((int*)(ws + WS_GS));
    ring_recur<<<NGROUP * NSLICE, TPB, 0, stream>>>(wcat, Agd, stg, gS, gC, gSn, gcnt,
                                                    out_rd7, out_r);
}
